// InfinityV2Model_36077725286429
// MI455X (gfx1250) — matrix operand order repaired and checked by reading, not compiled, not run on hardware
//
#include <hip/hip_runtime.h>
#include <stdint.h>

// ---------------- model dims ----------------
#define B_     2
#define T_     1024
#define D_     1024
#define H_     16
#define HD_    64
#define L_     4
#define DFF_   4096
#define VOCAB_ 16384
#define M_     (B_*T_)     // 2048 token rows

// ---------------- WMMA types ----------------
typedef __bf16 bf16_t;
typedef bf16_t v16bf __attribute__((ext_vector_type(16)));
typedef float  v8f   __attribute__((ext_vector_type(8)));

union Frag {
  uint4 q[2];
  v16bf v;
  unsigned short us[16];
};

__device__ __forceinline__ unsigned short f2bf(float f) {
  unsigned u = __float_as_uint(f);
  u += 0x7FFFu + ((u >> 16) & 1u);          // round-to-nearest-even
  return (unsigned short)(u >> 16);
}

__device__ __forceinline__ v8f wmma_bf16(const v16bf& a, const v16bf& b, const v8f& c) {
  // D = A(16x32) * B(32x16) + C, f32 accumulate
  return __builtin_amdgcn_wmma_f32_16x16x32_bf16(false, a, false, b, (short)0, c,
                                                 false, false);
}

__device__ __forceinline__ v8f vzero8() {
  v8f z;
#pragma unroll
  for (int i = 0; i < 8; ++i) z[i] = 0.f;
  return z;
}

// ---- CDNA5 async global->LDS copy (ASYNCcnt-tracked, no VGPR round trip) ----
// each lane copies 16 bytes from gptr to LDS byte offset lds_off
__device__ __forceinline__ void async_ld_b128(unsigned lds_off, const void* gptr) {
  asm volatile("global_load_async_to_lds_b128 %0, %1, off"
               :: "v"(lds_off), "v"(gptr)
               : "memory");
}
#define WAIT_ASYNC(n) asm volatile("s_wait_asynccnt " #n ::: "memory")

// low 32 bits of a generic pointer into the LDS aperture == LDS byte offset
__device__ __forceinline__ unsigned lds_off_of(const void* p) {
  return (unsigned)(size_t)p;
}

// ---------------- elementwise kernels ----------------
__global__ void cast_f32_bf16_kernel(const float* __restrict__ src,
                                     unsigned short* __restrict__ dst, long n) {
  long i = (long)blockIdx.x * blockDim.x + threadIdx.x;
  long stride = (long)gridDim.x * blockDim.x;
  for (; i < n; i += stride) dst[i] = f2bf(src[i]);
}

__global__ void embed_gather_kernel(const int* __restrict__ tok,
                                    const float* __restrict__ emb,
                                    float* __restrict__ x) {
  int idx = blockIdx.x * blockDim.x + threadIdx.x;   // M_*D_ threads
  int row = idx >> 10, col = idx & (D_ - 1);
  x[idx] = emb[(size_t)tok[row] * D_ + col];
}

__global__ __launch_bounds__(256)
void rmsnorm_kernel(const float* __restrict__ X, const float* __restrict__ W,
                    unsigned short* __restrict__ Y) {
  int row = blockIdx.x;
  const float* x = X + (size_t)row * D_;
  float ss = 0.f;
  for (int i = threadIdx.x; i < D_; i += 256) { float v = x[i]; ss += v * v; }
  for (int m = 1; m < 32; m <<= 1) ss += __shfl_xor(ss, m, 32);
  __shared__ float red[8];
  __shared__ float stot;
  int wv = threadIdx.x >> 5, ln = threadIdx.x & 31;
  if (ln == 0) red[wv] = ss;
  __syncthreads();
  if (threadIdx.x < 8) {
    float v = red[threadIdx.x];
    for (int m = 1; m < 8; m <<= 1) v += __shfl_xor(v, m, 32);
    if (threadIdx.x == 0) stot = v;
  }
  __syncthreads();
  float rinv = rsqrtf(stot * (1.f / D_) + 1e-6f);
  for (int i = threadIdx.x; i < D_; i += 256)
    Y[(size_t)row * D_ + i] = f2bf(x[i] * rinv * W[i]);
}

// q/k: complex rotate on (even,odd) pairs; output bf16
__global__ void rope_kernel(const float* __restrict__ Qf, const float* __restrict__ Kf,
                            const int* __restrict__ pos,
                            unsigned short* __restrict__ Qb, unsigned short* __restrict__ Kb) {
  int idx = blockIdx.x * blockDim.x + threadIdx.x;   // B*T*H*(HD/2) threads
  int i = idx & 31;                 // pair index 0..31
  int t = (idx >> 9) & (T_ - 1);
  int b = idx >> 19;
  int p = pos[b * T_ + t];
  float inv = __powf(10000.f, -(float)(2 * i) * (1.f / HD_));
  float ang = (float)p * inv;
  float s, c;
  __sincosf(ang, &s, &c);
  size_t base = (size_t)idx * 2;    // [b,t,h,2i]
  float qr = Qf[base], qi = Qf[base + 1];
  Qb[base]     = f2bf(qr * c - qi * s);
  Qb[base + 1] = f2bf(qr * s + qi * c);
  float kr = Kf[base], ki = Kf[base + 1];
  Kb[base]     = f2bf(kr * c - ki * s);
  Kb[base + 1] = f2bf(kr * s + ki * c);
}

__global__ void swiglu_kernel(const float* __restrict__ g1, const float* __restrict__ g3,
                              unsigned short* __restrict__ g) {
  int idx = blockIdx.x * blockDim.x + threadIdx.x;   // M_*DFF_ threads
  float a = g1[idx];
  float sig = 1.f / (1.f + __expf(-a));
  g[idx] = f2bf(a * sig * g3[idx]);
}

// ---------------- WMMA GEMM: C[M,N] = A[M,K](bf16) * W[N,K]^T (+Res) ----------------
// Double-buffered LDS tiles filled by global_load_async_to_lds_b128; the next
// K-tile streams in while WMMAs consume the current one.
#define BM 128
#define BN 128
#define BK 64          // two 32-wide WMMA k-steps per tile

__global__ __launch_bounds__(256)
void gemm_bf16_wmma_kernel(const unsigned short* __restrict__ A,
                           const unsigned short* __restrict__ Bw,
                           const float* __restrict__ Res,
                           float* __restrict__ C,
                           int M, int N, int K) {
  __shared__ __align__(16) unsigned short sA[2][BM * BK];   // 2 x 16KB
  __shared__ __align__(16) unsigned short sB[2][BN * BK];   // 2 x 16KB
  int tid = threadIdx.x;
  int wv = tid >> 5, ln = tid & 31;
  int hf = ln >> 4, l15 = ln & 15;
  int bm = blockIdx.y * BM, bn = blockIdx.x * BN;
  int wm = (wv & 3) * 32;        // 4 waves along M
  int wn = (wv >> 2) * 64;       // 2 waves along N

  v8f acc[2][4];
#pragma unroll
  for (int i = 0; i < 2; ++i)
#pragma unroll
    for (int j = 0; j < 4; ++j) acc[i][j] = vzero8();

  // stage one 128x64 bf16 tile of A and of W into LDS buffer `buf`
  auto stage = [&](int buf, int k0) {
#pragma unroll
    for (int i = 0; i < 4; ++i) {
      int v = tid + i * 256;               // 1024 vec8 per matrix tile
      int row = v >> 3, c8 = (v & 7) * 8;
      async_ld_b128(lds_off_of(&sA[buf][row * BK + c8]),
                    &A[(size_t)(bm + row) * K + k0 + c8]);
      async_ld_b128(lds_off_of(&sB[buf][row * BK + c8]),
                    &Bw[(size_t)(bn + row) * K + k0 + c8]);
    }
  };

  int nt = K / BK;
  int cur = 0;
  stage(0, 0);

  for (int kt = 0; kt < nt; ++kt) {
    if (kt + 2 < nt) {   // lead L2 with a prefetch two tiles ahead
      __builtin_prefetch(&A[(size_t)(bm + (tid >> 1)) * K + (kt + 2) * BK + (tid & 1) * 32], 0, 1);
      __builtin_prefetch(&Bw[(size_t)(bn + (tid >> 1)) * K + (kt + 2) * BK + (tid & 1) * 32], 0, 1);
    }
    if (kt + 1 < nt) {
      stage(cur ^ 1, (kt + 1) * BK);       // 8 more async loads in flight
      WAIT_ASYNC(0x8);                     // current tile done; next still flying
    } else {
      WAIT_ASYNC(0x0);
    }
    __syncthreads();

    const unsigned short* a = sA[cur];
    const unsigned short* b = sB[cur];
#pragma unroll
    for (int ks = 0; ks < 2; ++ks) {
      Frag af[2], bfr[4];
      // A fragment: lane row = l15; ks of (hf*8..+7, 16+hf*8..+7)
#pragma unroll
      for (int i = 0; i < 2; ++i) {
        int r = (wm + i * 16 + l15) * BK + ks * 32;
        af[i].q[0] = *(const uint4*)&a[r + hf * 8];
        af[i].q[1] = *(const uint4*)&a[r + 16 + hf * 8];
      }
      // B fragment: lane col = l15; ks of (hf*8..+7, 16+hf*8..+7)
#pragma unroll
      for (int j = 0; j < 4; ++j) {
        int r = (wn + j * 16 + l15) * BK + ks * 32;
        bfr[j].q[0] = *(const uint4*)&b[r + hf * 8];
        bfr[j].q[1] = *(const uint4*)&b[r + 16 + hf * 8];
      }
#pragma unroll
      for (int i = 0; i < 2; ++i)
#pragma unroll
        for (int j = 0; j < 4; ++j)
          acc[i][j] = wmma_bf16(af[i].v, bfr[j].v, acc[i][j]);
    }
    __syncthreads();
    cur ^= 1;
  }

  // C/D layout: VGPR r -> row r + hf*8, col = l15
#pragma unroll
  for (int i = 0; i < 2; ++i)
#pragma unroll
    for (int j = 0; j < 4; ++j)
#pragma unroll
      for (int r = 0; r < 8; ++r) {
        int m = bm + wm + i * 16 + r + hf * 8;
        int n = bn + wn + j * 16 + l15;
        float v = acc[i][j][r];
        if (Res) v += Res[(size_t)m * N + n];
        C[(size_t)m * N + n] = v;
      }
}

// ---------------- flash attention (no mask), bf16 in/out ----------------
// Q,K,V,O layout: [B,T,H,HD] bf16. One block per (b,h, 128 q-rows); 8 waves,
// each wave owns 16 queries; stream keys in chunks of 32. K chunk staged via
// async LDS loads; V staged transposed (element scatter) for contiguous
// B-fragment reads.
__global__ __launch_bounds__(256)
void flash_attn_kernel(const unsigned short* __restrict__ Q,
                       const unsigned short* __restrict__ Kk,
                       const unsigned short* __restrict__ V,
                       unsigned short* __restrict__ O) {
  __shared__ __align__(16) unsigned short sK[32 * 64];   // [s][hd]
  __shared__ __align__(16) unsigned short sV[64 * 32];   // transposed [hd][s]
  __shared__ __align__(16) float sP[8][16 * 32];         // per-wave P staging
  int tid = threadIdx.x, wv = tid >> 5, ln = tid & 31;
  int hf = ln >> 4, l15 = ln & 15;
  int bh = blockIdx.x >> 3;            // T_/128 = 8 q-tiles per (b,h)
  int b = bh >> 4, h = bh & (H_ - 1);
  int q0 = (blockIdx.x & 7) * 128 + wv * 16;
  const float scale = 0.125f;          // 1/sqrt(64)

  Frag qf[2];
  {
    size_t base = (((size_t)b * T_ + q0 + l15) * H_ + h) * HD_;
#pragma unroll
    for (int j = 0; j < 2; ++j) {      // hd 0..31 / 32..63
      qf[j].q[0] = *(const uint4*)&Q[base + j * 32 + hf * 8];
      qf[j].q[1] = *(const uint4*)&Q[base + j * 32 + 16 + hf * 8];
    }
  }

  v8f o[4];
#pragma unroll
  for (int f = 0; f < 4; ++f) o[f] = vzero8();
  float rm[8], rl[8];
#pragma unroll
  for (int r = 0; r < 8; ++r) { rm[r] = -1e30f; rl[r] = 0.f; }

  int srow = tid >> 3, sc8 = (tid & 7) * 8;
  for (int s0 = 0; s0 < T_; s0 += 32) {
    // K chunk: async straight into LDS; V chunk: registers + transpose scatter
    size_t g = (((size_t)b * T_ + s0 + srow) * H_ + h) * HD_ + sc8;
    async_ld_b128(lds_off_of(&sK[srow * 64 + sc8]), &Kk[g]);
    unsigned short tmp[8];
    *(uint4*)tmp = *(const uint4*)&V[g];
#pragma unroll
    for (int e = 0; e < 8; ++e) sV[(sc8 + e) * 32 + srow] = tmp[e];
    WAIT_ASYNC(0x0);
    __syncthreads();

    // scores: S(16x32) = Q(16x64) * K_chunk^T(64x32), two 16-wide n-tiles
    v8f c0 = vzero8(), c1 = vzero8();
#pragma unroll
    for (int kk = 0; kk < 2; ++kk) {
      Frag kf;
      int off = kk * 32 + hf * 8;
      kf.q[0] = *(const uint4*)&sK[l15 * 64 + off];
      kf.q[1] = *(const uint4*)&sK[l15 * 64 + off + 16];
      c0 = wmma_bf16(qf[kk].v, kf.v, c0);
      kf.q[0] = *(const uint4*)&sK[(16 + l15) * 64 + off];
      kf.q[1] = *(const uint4*)&sK[(16 + l15) * 64 + off + 16];
      c1 = wmma_bf16(qf[kk].v, kf.v, c1);
    }

    // online softmax per row (row lives in 16-lane half, per-VGPR r)
#pragma unroll
    for (int r = 0; r < 8; ++r) {
      float a0 = c0[r] * scale, a1 = c1[r] * scale;
      float mx = fmaxf(a0, a1);
      for (int m = 1; m < 16; m <<= 1) mx = fmaxf(mx, __shfl_xor(mx, m, 32));
      float nm = fmaxf(rm[r], mx);
      float al = __expf(rm[r] - nm);
      rm[r] = nm;
      float p0 = __expf(a0 - nm), p1 = __expf(a1 - nm);
      float sum = p0 + p1;
      for (int m = 1; m < 16; m <<= 1) sum += __shfl_xor(sum, m, 32);
      rl[r] = rl[r] * al + sum;
      int mrow = r + hf * 8;
      sP[wv][mrow * 32 + l15] = p0;
      sP[wv][mrow * 32 + 16 + l15] = p1;
#pragma unroll
      for (int f = 0; f < 4; ++f) o[f][r] *= al;
    }

    // re-layout P (C-frag f32 in LDS) -> A-frag bf16
    Frag pf;
    {
      float f0[16];
      int mb = l15 * 32;
      *(float4*)&f0[0]  = *(const float4*)&sP[wv][mb + hf * 8];
      *(float4*)&f0[4]  = *(const float4*)&sP[wv][mb + hf * 8 + 4];
      *(float4*)&f0[8]  = *(const float4*)&sP[wv][mb + 16 + hf * 8];
      *(float4*)&f0[12] = *(const float4*)&sP[wv][mb + 16 + hf * 8 + 4];
#pragma unroll
      for (int e = 0; e < 16; ++e) pf.us[e] = f2bf(f0[e]);
    }

    // O(16x64) += P(16x32) * V(32x64) : 4 n-tiles
#pragma unroll
    for (int f = 0; f < 4; ++f) {
      Frag vf;
      int nb = (f * 16 + l15) * 32 + hf * 8;
      vf.q[0] = *(const uint4*)&sV[nb];
      vf.q[1] = *(const uint4*)&sV[nb + 16];
      o[f] = wmma_bf16(pf.v, vf.v, o[f]);
    }
    __syncthreads();
  }

  // normalize and write O as bf16 [b,t,h,hd]
#pragma unroll
  for (int f = 0; f < 4; ++f)
#pragma unroll
    for (int r = 0; r < 8; ++r) {
      int t = q0 + r + hf * 8;
      int hd = f * 16 + l15;
      O[(((size_t)b * T_ + t) * H_ + h) * HD_ + hd] = f2bf(o[f][r] / rl[r]);
    }
}

// ---------------- host driver ----------------
extern "C" void kernel_launch(void* const* d_in, const int* in_sizes, int n_in,
                              void* d_out, int out_size, void* d_ws, size_t ws_size,
                              hipStream_t stream) {
  (void)in_sizes; (void)n_in; (void)out_size; (void)ws_size;
  const int*   tok     = (const int*)d_in[0];
  const int*   pos     = (const int*)d_in[1];
  const float* embed   = (const float*)d_in[2];
  const float* attn_nw = (const float*)d_in[3];
  const float* wq      = (const float*)d_in[4];
  const float* wk      = (const float*)d_in[5];
  const float* wv      = (const float*)d_in[6];
  const float* wo      = (const float*)d_in[7];
  const float* ff_nw   = (const float*)d_in[8];
  const float* w1      = (const float*)d_in[9];
  const float* w2      = (const float*)d_in[10];
  const float* w3      = (const float*)d_in[11];
  const float* norm_w  = (const float*)d_in[12];
  float* logits = (float*)d_out;                    // [B,T,VOCAB] f32

  // deterministic workspace carve (~300 MB)
  char* p = (char*)d_ws;
  auto carve = [&](size_t bytes) -> void* {
    void* r = (void*)p;
    p += (bytes + 255) & ~(size_t)255;
    return r;
  };
  const size_t DD = (size_t)D_ * D_, FD = (size_t)DFF_ * D_;
  unsigned short* wq_b  = (unsigned short*)carve(L_ * DD * 2);
  unsigned short* wk_b  = (unsigned short*)carve(L_ * DD * 2);
  unsigned short* wv_b  = (unsigned short*)carve(L_ * DD * 2);
  unsigned short* wo_b  = (unsigned short*)carve(L_ * DD * 2);
  unsigned short* w1_b  = (unsigned short*)carve(L_ * FD * 2);
  unsigned short* w2_b  = (unsigned short*)carve(L_ * FD * 2);
  unsigned short* w3_b  = (unsigned short*)carve(L_ * FD * 2);
  unsigned short* emb_b = (unsigned short*)carve((size_t)VOCAB_ * D_ * 2);
  float*          x     = (float*)carve((size_t)M_ * D_ * 4);
  unsigned short* hbuf  = (unsigned short*)carve((size_t)M_ * D_ * 2);
  float*          qf    = (float*)carve((size_t)M_ * D_ * 4);
  float*          kf    = (float*)carve((size_t)M_ * D_ * 4);
  float*          vf    = (float*)carve((size_t)M_ * D_ * 4);
  unsigned short* qb    = (unsigned short*)carve((size_t)M_ * D_ * 2);
  unsigned short* kb    = (unsigned short*)carve((size_t)M_ * D_ * 2);
  unsigned short* vb    = (unsigned short*)carve((size_t)M_ * D_ * 2);
  unsigned short* ob    = (unsigned short*)carve((size_t)M_ * D_ * 2);
  float*          g1    = (float*)carve((size_t)M_ * DFF_ * 4);
  float*          g3    = (float*)carve((size_t)M_ * DFF_ * 4);
  unsigned short* gb    = (unsigned short*)carve((size_t)M_ * DFF_ * 2);
  unsigned short* xn    = (unsigned short*)carve((size_t)M_ * D_ * 2);

  auto cast = [&](const float* s, unsigned short* d, long n) {
    int blocks = (int)((n + 256L * 8 - 1) / (256L * 8));
    cast_f32_bf16_kernel<<<blocks, 256, 0, stream>>>(s, d, n);
  };
  auto gemm = [&](const unsigned short* A, const unsigned short* Bw,
                  const float* Res, float* C, int M, int N, int K) {
    dim3 grid(N / BN, M / BM);
    gemm_bf16_wmma_kernel<<<grid, 256, 0, stream>>>(A, Bw, Res, C, M, N, K);
  };

  // one-time weight conversion to bf16 (L2-resident afterwards)
  cast(embed, emb_b, (long)VOCAB_ * D_);
  cast(wq, wq_b, (long)L_ * DD);
  cast(wk, wk_b, (long)L_ * DD);
  cast(wv, wv_b, (long)L_ * DD);
  cast(wo, wo_b, (long)L_ * DD);
  cast(w1, w1_b, (long)L_ * FD);
  cast(w2, w2_b, (long)L_ * FD);
  cast(w3, w3_b, (long)L_ * FD);

  embed_gather_kernel<<<(M_ * D_) / 256, 256, 0, stream>>>(tok, embed, x);

  const int ropeN  = B_ * T_ * H_ * (HD_ / 2);      // one thread per rotated pair
  const int attnG  = B_ * H_ * (T_ / 128);

  for (int l = 0; l < L_; ++l) {
    // ---- attention block ----
    rmsnorm_kernel<<<M_, 256, 0, stream>>>(x, attn_nw + (size_t)l * D_, hbuf);
    gemm(hbuf, wq_b + (size_t)l * DD, nullptr, qf, M_, D_, D_);
    gemm(hbuf, wk_b + (size_t)l * DD, nullptr, kf, M_, D_, D_);
    gemm(hbuf, wv_b + (size_t)l * DD, nullptr, vf, M_, D_, D_);
    rope_kernel<<<ropeN / 256, 256, 0, stream>>>(qf, kf, pos, qb, kb);
    cast(vf, vb, (long)M_ * D_);
    flash_attn_kernel<<<attnG, 256, 0, stream>>>(qb, kb, vb, ob);
    gemm(ob, wo_b + (size_t)l * DD, x, x, M_, D_, D_);    // x += o @ Wo^T

    // ---- SwiGLU FFN ----
    rmsnorm_kernel<<<M_, 256, 0, stream>>>(x, ff_nw + (size_t)l * D_, hbuf);
    gemm(hbuf, w1_b + (size_t)l * FD, nullptr, g1, M_, DFF_, D_);
    gemm(hbuf, w3_b + (size_t)l * FD, nullptr, g3, M_, DFF_, D_);
    swiglu_kernel<<<(M_ * DFF_) / 256, 256, 0, stream>>>(g1, g3, gb);
    gemm(gb, w2_b + (size_t)l * FD, x, x, M_, D_, DFF_);  // x += g @ W2^T
  }

  // final norm + tied LM head
  rmsnorm_kernel<<<M_, 256, 0, stream>>>(x, norm_w, xn);
  gemm(xn, emb_b, nullptr, logits, M_, VOCAB_, D_);
}
